// SlotDecoder_28501402976981
// MI455X (gfx1250) — matrix operand order repaired and checked by reading, not compiled, not run on hardware
//
#include <hip/hip_runtime.h>
#include <hip/hip_bf16.h>
#include <math.h>

#define DEVFN __device__ __forceinline__

typedef __attribute__((ext_vector_type(16))) _Float16 v16h;
typedef __attribute__((ext_vector_type(8)))  float    v8f;
typedef __attribute__((ext_vector_type(4)))  unsigned int u32x4;
typedef __attribute__((ext_vector_type(8)))  int      i32x8;
typedef __attribute__((ext_vector_type(4)))  int      i32x4;

// Problem dims (fixed by reference)
constexpr int B_ = 64, S_ = 512, H_ = 768, F_ = 2048, T_ = 128;
constexpr int BS = B_ * S_;
constexpr float EPS   = 1e-6f;
constexpr float SLOPE = 0.01f;

// GEMM tiling: block = 256 threads = 8 waves (2 M-waves x 4 N-waves)
constexpr int TM = 128, TN = 64, TK = 32;
constexpr int LDA = 40;  // halfword row stride: 64B row + 16B pad => 80B, conflict-free, 16B aligned
constexpr int LDB = 40;

// Use the Tensor Data Mover when the toolchain exposes it (device pass only).
#if defined(__HIP_DEVICE_COMPILE__) && defined(__has_builtin)
#  if __has_builtin(__builtin_amdgcn_tensor_load_to_lds)
#    define USE_TDM 1
#  endif
#endif
#ifndef USE_TDM
#  define USE_TDM 0
#endif

// ---------------------------------------------------------------------------
// Weight transpose + f32->f16 convert: in (R,C) row-major -> out (C,R)
// ---------------------------------------------------------------------------
__global__ void transpose_to_f16(const float* __restrict__ in, _Float16* __restrict__ out,
                                 int R, int C) {
  long long o = (long long)blockIdx.x * blockDim.x + threadIdx.x;
  if (o >= (long long)R * C) return;
  int r = (int)(o % R);
  long long c = o / R;
  out[o] = (_Float16)in[(long long)r * C + c];
}

// ---------------------------------------------------------------------------
// RMSNorm: one block per token row (768 elems, 3 per thread), f16 output
// ---------------------------------------------------------------------------
__global__ __launch_bounds__(256)
void rmsnorm_kernel(const float* __restrict__ x, const float* __restrict__ w,
                    _Float16* __restrict__ h) {
  const int row = blockIdx.x;
  const int tid = threadIdx.x;
  const float* xr = x + (size_t)row * H_;
  const float v0 = xr[tid], v1 = xr[tid + 256], v2 = xr[tid + 512];
  __shared__ float red[256];
  red[tid] = v0 * v0 + v1 * v1 + v2 * v2;
  __syncthreads();
  for (int o = 128; o > 0; o >>= 1) {
    if (tid < o) red[tid] += red[tid + o];
    __syncthreads();
  }
  const float scale = rsqrtf(red[0] * (1.0f / H_) + EPS);
  _Float16* hr = h + (size_t)row * H_;
  hr[tid]       = (_Float16)(v0 * scale * w[tid]);
  hr[tid + 256] = (_Float16)(v1 * scale * w[tid + 256]);
  hr[tid + 512] = (_Float16)(v2 * scale * w[tid + 512]);
}

// ---------------------------------------------------------------------------
// WMMA fragment loaders (CDNA5 16-bit layouts, wave32)
// A 16x32: lane m=lane%16; elems 0..7 K=8*(lane/16)+0..7; elems 8..15 K=+16
// B 32x16 (staged transposed [n][k]): lane n=lane%16; elems 0..7 K=8*(lane/16)+0..7; elems 8..15 K=+16
// ---------------------------------------------------------------------------
DEVFN v16h load_frag_a(const _Float16* sA, int row, int lane) {
  const _Float16* p = sA + row * LDA + ((lane >> 4) << 3);
  union { v16h v; uint4 q[2]; } u;
  u.q[0] = *(const uint4*)p;
  u.q[1] = *(const uint4*)(p + 16);
  return u.v;
}
DEVFN v16h load_frag_b(const _Float16* sB, int row, int lane) {
  const _Float16* p = sB + row * LDB + ((lane >> 4) << 3);
  union { v16h v; uint4 q[2]; } u;
  u.q[0] = *(const uint4*)p;
  u.q[1] = *(const uint4*)(p + 16);
  return u.v;
}

#if USE_TDM
// ---------------------------------------------------------------------------
// Tensor DMA descriptor builders (CDNA5 ISA)
// ---------------------------------------------------------------------------
DEVFN u32x4 tdm_g0(unsigned lds_addr, unsigned long long gaddr) {
  u32x4 g;
  g[0] = 1u;                                             // count=1 valid descriptor
  g[1] = lds_addr;                                       // LDS byte address
  g[2] = (unsigned)(gaddr & 0xffffffffu);                // global_addr[31:0]
  g[3] = (unsigned)((gaddr >> 32) & 0x1ffffffu)          // global_addr[56:32]
       | (2u << 30);                                     // type = 2 ("image")
  return g;
}
// 2-D tile of 16-bit elements, LDS padding: 16 DWORDs data then 4 DWORDs pad
// (matches 32-halfword rows stored at LDA/LDB = 40-halfword stride).
DEVFN i32x8 tdm_g1(unsigned tensor_d0, unsigned tensor_d1,
                   unsigned tile_d0, unsigned tile_d1,
                   unsigned long long stride0) {
  union { i32x8 v; unsigned long long q[4]; } u;
  const unsigned flags = (1u << 16)    // data_size = 2 bytes
                       | (1u << 20)    // pad_enable
                       | (3u << 22)    // pad_interval: 16 DWORDs
                       | (3u << 25);   // pad_amount:   4 DWORDs
  u.q[0] = (unsigned long long)flags
         | ((unsigned long long)(tensor_d0 & 0xffffu) << 48);
  u.q[1] = (unsigned long long)(tensor_d0 >> 16)
         | ((unsigned long long)tensor_d1 << 16)
         | ((unsigned long long)(tile_d0 & 0xffffu) << 48);
  u.q[2] = (unsigned long long)(tile_d1 & 0xffffu)
         | ((stride0 & 0xffffffffull) << 32);
  u.q[3] = (stride0 >> 32) & 0xffffull;
  return u.v;
}
DEVFN void tdm_load(u32x4 g0, i32x8 g1) {
  const i32x4 z4 = (i32x4){0, 0, 0, 0};
#if __clang_major__ >= 23
  const i32x8 z8 = (i32x8){0, 0, 0, 0, 0, 0, 0, 0};
  __builtin_amdgcn_tensor_load_to_lds(g0, g1, z4, z4, z8, 0);
#else
  __builtin_amdgcn_tensor_load_to_lds(g0, g1, z4, z4, 0);
#endif
}
#endif  // USE_TDM

// ---------------------------------------------------------------------------
// One K-step of WMMA compute from staged LDS tiles.
// ---------------------------------------------------------------------------
template <int MODE>
DEVFN void compute_step(const _Float16* cA, const _Float16* cB, const _Float16* cB2,
                        int wm, int nrow, int lane, v8f (&acc)[4], v8f (&acc2)[4]) {
  const v16h bf = load_frag_b(cB, nrow, lane);
  if constexpr (MODE == 0) {
    const v16h bf2 = load_frag_b(cB2, nrow, lane);
#pragma unroll
    for (int mt = 0; mt < 4; ++mt) {
      const v16h af = load_frag_a(cA, (wm << 6) + (mt << 4) + (lane & 15), lane);
      acc[mt]  = __builtin_amdgcn_wmma_f32_16x16x32_f16(false, af, false, bf,
                                                        (short)0, acc[mt], false, false);
      acc2[mt] = __builtin_amdgcn_wmma_f32_16x16x32_f16(false, af, false, bf2,
                                                        (short)0, acc2[mt], false, false);
    }
  } else {
#pragma unroll
    for (int mt = 0; mt < 4; ++mt) {
      const v16h af = load_frag_a(cA, (wm << 6) + (mt << 4) + (lane & 15), lane);
      acc[mt] = __builtin_amdgcn_wmma_f32_16x16x32_f16(false, af, false, bf,
                                                       (short)0, acc[mt], false, false);
    }
  }
}

// ---------------------------------------------------------------------------
// WMMA GEMM: C(M,N) = A(M,K) @ Bt(N,K)^T, f16 in / f32 accumulate.
// MODE 0: dual GEMM (Bt & Bt2 share A), out = f16( silu(C1) * C2 )   [w1/w3+SiLU]
// MODE 1: out = f16( C + aux[m,n] )                                   [w2 + residual]
// MODE 2: out = f32 leaky_relu( C + aux[n] )                          [fc_w + bias]
// Block tile 128x64, K-step 32. TDM path: double-buffered LDS, wave 0 issues
// tensor_load_to_lds for stage k+1 while all waves compute stage k.
// ---------------------------------------------------------------------------
template <int MODE>
__global__ __launch_bounds__(256)
void gemm_kernel(const _Float16* __restrict__ A,
                 const _Float16* __restrict__ Bt,
                 const _Float16* __restrict__ Bt2,
                 const float* __restrict__ aux,
                 void* __restrict__ out,
                 int K, int Ntot) {
  const int tid  = threadIdx.x;
  const int lane = tid & 31;
  const int w    = tid >> 5;
  const int wm   = w & 1;
  const int wn   = w >> 1;
  const int m0   = blockIdx.y * TM;
  const int n0   = blockIdx.x * TN;

  constexpr int ASZ = TM * LDA;   // halfwords per A stage
  constexpr int BSZ = TN * LDB;   // halfwords per B stage
  __shared__ __align__(16) _Float16 sA[2 * ASZ];
  __shared__ __align__(16) _Float16 sB[2 * BSZ];
  constexpr int NB2 = (MODE == 0) ? 2 * BSZ : 8;
  __shared__ __align__(16) _Float16 sB2[NB2];

  v8f acc[4], acc2[4];
#pragma unroll
  for (int i = 0; i < 4; ++i) {
    acc[i]  = (v8f){0.f, 0.f, 0.f, 0.f, 0.f, 0.f, 0.f, 0.f};
    acc2[i] = (v8f){0.f, 0.f, 0.f, 0.f, 0.f, 0.f, 0.f, 0.f};
  }

  const int nrow = (wn << 4) + (lane & 15);
  const int nK   = K / TK;

#if USE_TDM
  // ---- TDM double-buffered pipeline ----
  const unsigned ldsA  = (unsigned)(uintptr_t)(const void*)sA;
  const unsigned ldsB  = (unsigned)(uintptr_t)(const void*)sB;
  const unsigned ldsB2 = (unsigned)(uintptr_t)(const void*)sB2;
  const i32x8 g1A = tdm_g1((unsigned)K, 1u << 20, TK, TM, (unsigned long long)K);
  const i32x8 g1B = tdm_g1((unsigned)K, 1u << 20, TK, TN, (unsigned long long)K);

  auto issue_stage = [&](int buf, int k0) {
    tdm_load(tdm_g0(ldsA + (unsigned)buf * (ASZ * 2),
                    (unsigned long long)(uintptr_t)(A + (size_t)m0 * K + k0)), g1A);
    tdm_load(tdm_g0(ldsB + (unsigned)buf * (BSZ * 2),
                    (unsigned long long)(uintptr_t)(Bt + (size_t)n0 * K + k0)), g1B);
    if constexpr (MODE == 0)
      tdm_load(tdm_g0(ldsB2 + (unsigned)buf * (BSZ * 2),
                      (unsigned long long)(uintptr_t)(Bt2 + (size_t)n0 * K + k0)), g1B);
  };

  if (w == 0) issue_stage(0, 0);
  for (int kk = 0; kk < nK; ++kk) {
    const int buf = kk & 1;
    if (w == 0) {
      if (kk + 1 < nK) {
        issue_stage(buf ^ 1, (kk + 1) * TK);
        // N just-issued descriptors may remain outstanding => current stage done.
        if constexpr (MODE == 0) __builtin_amdgcn_s_wait_tensorcnt(3);
        else                     __builtin_amdgcn_s_wait_tensorcnt(2);
      } else {
        __builtin_amdgcn_s_wait_tensorcnt(0);
      }
    }
    __syncthreads();  // release compute waves once stage `buf` is in LDS
    compute_step<MODE>(sA + buf * ASZ, sB + buf * BSZ, sB2 + (MODE == 0 ? buf * BSZ : 0),
                       wm, nrow, lane, acc, acc2);
    __syncthreads();  // all waves done with `buf` before it becomes a DMA target
  }
#else
  // ---- fallback: manual global->LDS staging, single buffer ----
  for (int kk = 0; kk < nK; ++kk) {
    const int k0 = kk * TK;
    __syncthreads();
#pragma unroll
    for (int i = 0; i < 2; ++i) {
      const int cc  = tid + (i << 8);
      const int row = cc >> 2;
      const int col = (cc & 3) << 3;
      *(uint4*)(sA + row * LDA + col) =
          *(const uint4*)(A + (size_t)(m0 + row) * K + k0 + col);
    }
    {
      const int row = tid >> 2;
      const int col = (tid & 3) << 3;
      *(uint4*)(sB + row * LDB + col) =
          *(const uint4*)(Bt + (size_t)(n0 + row) * K + k0 + col);
      if constexpr (MODE == 0)
        *(uint4*)(sB2 + row * LDB + col) =
            *(const uint4*)(Bt2 + (size_t)(n0 + row) * K + k0 + col);
    }
    __syncthreads();
    compute_step<MODE>(sA, sB, sB2, wm, nrow, lane, acc, acc2);
  }
#endif

  // Epilogue. C layout: VGPR r -> m = r + 8*(lane/16); n = lane%16.
  const int mloc = (lane >> 4) << 3;
  const int gn   = n0 + (wn << 4) + (lane & 15);
#pragma unroll
  for (int mt = 0; mt < 4; ++mt) {
#pragma unroll
    for (int r = 0; r < 8; ++r) {
      const int gm = m0 + (wm << 6) + (mt << 4) + mloc + r;
      const float v = acc[mt][r];
      if constexpr (MODE == 0) {
        const float s = v / (1.f + __expf(-v));  // silu
        ((_Float16*)out)[(size_t)gm * Ntot + gn] = (_Float16)(s * acc2[mt][r]);
      } else if constexpr (MODE == 1) {
        ((_Float16*)out)[(size_t)gm * Ntot + gn] =
            (_Float16)(v + aux[(size_t)gm * Ntot + gn]);
      } else {
        float y = v + aux[gn];
        y = y > 0.f ? y : SLOPE * y;
        ((float*)out)[(size_t)gm * Ntot + gn] = y;
      }
    }
  }
}

// ---------------------------------------------------------------------------
// CRF: one block per batch element, T=128 threads. trans cached in LDS (f16).
// ---------------------------------------------------------------------------
__global__ __launch_bounds__(128)
void crf_kernel(const float* __restrict__ em,       // (B,S,T)
                const float* __restrict__ start_t,  // (T)
                const float* __restrict__ end_t,    // (T)
                const float* __restrict__ trans,    // (T,T)
                const int* __restrict__ tags,       // (B,S) int32
                const unsigned char* __restrict__ mask,  // (B,S) bool (1 byte)
                float* __restrict__ per_batch) {
  const int b = blockIdx.x, j = threadIdx.x;
  const float* emb = em + (size_t)b * S_ * T_;
  const int* tg = tags + (size_t)b * S_;
  const unsigned char* mk = mask + (size_t)b * S_;

  __shared__ _Float16 s_tr[T_ * T_];   // 32 KB
  __shared__ float s_alpha[T_];
  __shared__ float s_red[T_];
  __shared__ float s_len[T_];
  __shared__ float s_num;

  for (int i = j; i < T_ * T_; i += T_) s_tr[i] = (_Float16)trans[i];

  // ---- numerator: parallel partials over t = 1..S-1 ----
  float npart = 0.f;
  float lpart = (j == 0) ? (mk[0] ? 1.f : 0.f) : 0.f;
  for (int t = 1 + j; t < S_; t += T_) {
    if (mk[t]) {
      npart += trans[tg[t - 1] * T_ + tg[t]] + emb[(size_t)t * T_ + tg[t]];
      lpart += 1.f;
    }
  }
  s_red[j] = npart;
  s_len[j] = lpart;
  __syncthreads();
  for (int o = 64; o > 0; o >>= 1) {
    if (j < o) { s_red[j] += s_red[j + o]; s_len[j] += s_len[j + o]; }
    __syncthreads();
  }
  if (j == 0) {
    const int len = (int)(s_len[0] + 0.5f);
    float num = s_red[0] + start_t[tg[0]] + emb[tg[0]];
    num += end_t[tg[len - 1]];
    s_num = num;
  }

  // ---- forward algorithm (online logsumexp) ----
  s_alpha[j] = start_t[j] + emb[j];
  __syncthreads();
  for (int t = 1; t < S_; ++t) {
    const float e = emb[(size_t)t * T_ + j];
    float M = -3.4e38f, Ssum = 0.f;
#pragma unroll 4
    for (int i = 0; i < T_; ++i) {
      const float v = s_alpha[i] + (float)s_tr[i * T_ + j];
      if (v > M) { Ssum = Ssum * __expf(M - v) + 1.f; M = v; }
      else       { Ssum += __expf(v - M); }
    }
    const float nxt  = M + __logf(Ssum) + e;
    const float aold = s_alpha[j];
    const float anew = mk[t] ? nxt : aold;
    __syncthreads();
    s_alpha[j] = anew;
    __syncthreads();
  }

  // ---- logZ = LSE_j(alpha[j] + end_t[j]) ----
  const float vj = s_alpha[j] + end_t[j];
  s_red[j] = vj;
  __syncthreads();
  for (int o = 64; o > 0; o >>= 1) {
    if (j < o) s_red[j] = fmaxf(s_red[j], s_red[j + o]);
    __syncthreads();
  }
  const float mx = s_red[0];
  __syncthreads();
  s_red[j] = __expf(vj - mx);
  __syncthreads();
  for (int o = 64; o > 0; o >>= 1) {
    if (j < o) s_red[j] += s_red[j + o];
    __syncthreads();
  }
  if (j == 0) per_batch[b] = s_num - (mx + __logf(s_red[0]));
}

__global__ void finalize_kernel(const float* __restrict__ pb, float* __restrict__ out) {
  __shared__ float r[64];
  const int j = threadIdx.x;
  r[j] = pb[j];
  __syncthreads();
  for (int o = 32; o > 0; o >>= 1) {
    if (j < o) r[j] += r[j + o];
    __syncthreads();
  }
  if (j == 0) out[0] = -r[0] * (1.f / B_);
}

// ---------------------------------------------------------------------------
extern "C" void kernel_launch(void* const* d_in, const int* in_sizes, int n_in,
                              void* d_out, int out_size, void* d_ws, size_t ws_size,
                              hipStream_t stream) {
  const float* x       = (const float*)d_in[0];   // (B,S,H)
  const float* rms_w   = (const float*)d_in[1];   // (H)
  const float* w1      = (const float*)d_in[2];   // (H,F)
  const float* w3      = (const float*)d_in[3];   // (H,F)
  const float* w2      = (const float*)d_in[4];   // (F,H)
  const float* fc_w    = (const float*)d_in[5];   // (H,T)
  const float* fc_b    = (const float*)d_in[6];   // (T)
  const float* start_t = (const float*)d_in[7];   // (T)
  const float* end_t   = (const float*)d_in[8];   // (T)
  const float* trans   = (const float*)d_in[9];   // (T,T)
  const int*   tags    = (const int*)d_in[10];    // (B,S)
  const unsigned char* mask = (const unsigned char*)d_in[11];  // (B,S) bool

  char* ws = (char*)d_ws;
  size_t off = 0;
  auto carve = [&](size_t bytes) {
    void* p = ws + off;
    off += (bytes + 255) & ~(size_t)255;
    return p;
  };
  _Float16* h16  = (_Float16*)carve((size_t)BS * H_ * 2);
  _Float16* w1t  = (_Float16*)carve((size_t)H_ * F_ * 2);  // (F,H) K-contig
  _Float16* w3t  = (_Float16*)carve((size_t)H_ * F_ * 2);  // (F,H)
  _Float16* w2t  = (_Float16*)carve((size_t)H_ * F_ * 2);  // (H,F)
  _Float16* fcwt = (_Float16*)carve((size_t)H_ * T_ * 2);  // (T,H)
  _Float16* g16  = (_Float16*)carve((size_t)BS * F_ * 2);
  _Float16* xp16 = (_Float16*)carve((size_t)BS * H_ * 2);
  float*    em   = (float*)carve((size_t)BS * T_ * 4);
  float*    pb   = (float*)carve((size_t)B_ * 4);

  const int nHF = H_ * F_;
  transpose_to_f16<<<(nHF + 255) / 256, 256, 0, stream>>>(w1, w1t, H_, F_);
  transpose_to_f16<<<(nHF + 255) / 256, 256, 0, stream>>>(w3, w3t, H_, F_);
  transpose_to_f16<<<(nHF + 255) / 256, 256, 0, stream>>>(w2, w2t, F_, H_);
  transpose_to_f16<<<(H_ * T_ + 255) / 256, 256, 0, stream>>>(fc_w, fcwt, H_, T_);

  rmsnorm_kernel<<<BS, 256, 0, stream>>>(x, rms_w, h16);

  dim3 g1(F_ / TN, BS / TM);
  gemm_kernel<0><<<g1, 256, 0, stream>>>(h16, w1t, w3t, nullptr, g16, H_, F_);

  dim3 g2(H_ / TN, BS / TM);
  gemm_kernel<1><<<g2, 256, 0, stream>>>(g16, w2t, nullptr, x, xp16, F_, H_);

  dim3 g3(T_ / TN, BS / TM);
  gemm_kernel<2><<<g3, 256, 0, stream>>>(xp16, fcwt, nullptr, fc_b, em, H_, T_);

  crf_kernel<<<B_, T_, 0, stream>>>(em, start_t, end_t, trans, tags, mask, pb);
  finalize_kernel<<<1, 64, 0, stream>>>(pb, (float*)d_out);
}
